// DeformableTemporalAttention_5961414607490
// MI455X (gfx1250) — matrix operand order repaired and checked by reading, not compiled, not run on hardware
//
#include <hip/hip_runtime.h>
#include <hip/hip_bf16.h>
#include <math.h>

// ---------------------------------------------------------------------------
// DeformableTemporalAttention for MI455X (gfx1250, wave32, WMMA)
// (B,N,C,H,W)=(2,4,256,64,64), T=64, K=8, P=6, D=32
// ---------------------------------------------------------------------------
#define BB   2
#define NN   4
#define CC   256
#define HH   64
#define WW   64
#define HWp  4096
#define TT   64
#define KK   8
#define PP   6
#define DD   32

typedef __attribute__((ext_vector_type(16))) __bf16 v16bf;
typedef __attribute__((ext_vector_type(8)))  float  v8f;

struct Bits256 { uint4 lo; uint4 hi; };

__device__ __forceinline__ unsigned short f2bf(float f) {
  unsigned int u = __float_as_uint(f);
  u += 0x7fffu + ((u >> 16) & 1u);          // round-to-nearest-even
  return (unsigned short)(u >> 16);
}

__device__ __forceinline__ float gelu_f(float x) {
  return 0.5f * x * (1.f + erff(x * 0.70710678118654752f));
}

#define EP_NONE 0
#define EP_GELU 1
#define EP_GATE 2

// ---------------------------------------------------------------------------
// Generic 1x1-conv GEMM: Y[o,pix] = sum_c W[o,c] * X[c,pix] + bias[o]
// bf16 WMMA 16x16x32, f32 accumulate. 256 threads = 8 waves; 64(M)x32(N) tile,
// K-step 64 (two WMMAs per staging phase, half the barriers).
// vt_store==1 writes Y as (batch, o/32, pix, o%32)   [channel-last per head]
// ---------------------------------------------------------------------------
__global__ __launch_bounds__(256)
void gemm1x1_wmma(const float* __restrict__ X, const float* __restrict__ Wt,
                  const float* __restrict__ bias, const float* __restrict__ gate,
                  float* __restrict__ Y,
                  int Cout, int Cin, int Npix,
                  long xbs, long xcs, long xps,
                  long ybs, long yos, long yps,
                  int epilogue, int vt_store)
{
  __shared__ alignas(16) unsigned short As[64][72];   // [m][k], 144B rows (16B mult)
  __shared__ alignas(16) unsigned short Bs[32][72];   // [col][k] (transposed)

  const int tid  = threadIdx.x;
  const int lane = tid & 31;
  const int wave = tid >> 5;
  const int wm   = wave >> 1;          // 0..3  (M sub-tile)
  const int wn   = wave & 1;           // 0..1  (N sub-tile)
  const int m0   = blockIdx.x * 64;
  const int n0   = blockIdx.y * 32;
  const int batch = blockIdx.z;

  const float* Xb = X + (long)batch * xbs;

  v8f acc = {};

  // staging decomposition
  const int aRow = tid >> 2;           // 0..63
  const int aK0  = (tid & 3) * 16;     // 0,16,32,48
  const int bKp  = (tid >> 3) * 2;     // 0,2,...,62 (pair of K rows)
  const int bC0  = (tid & 7) * 4;      // 0..28

  // fragment-load constants (CDNA5 16-bit A / B VGPR layouts)
  const int rA   = wm * 16 + (lane & 15);
  const int kloA = (lane >= 16) ? 8 : 0;
  const int colB = wn * 16 + (lane & 15);
  const int kbB  = (lane >= 16) ? 8 : 0;

  for (int kk = 0; kk < Cin; kk += 64) {
    // ---- stage A panel (64m x 64k) ----
    {
      int o = m0 + aRow;
      if (o < Cout) {
        const float* src = Wt + (long)o * Cin + kk + aK0;
        #pragma unroll
        for (int i = 0; i < 16; ++i) As[aRow][aK0 + i] = f2bf(src[i]);
      } else {
        #pragma unroll
        for (int i = 0; i < 16; ++i) As[aRow][aK0 + i] = 0;
      }
    }
    // ---- stage B panel (64k x 32col), transposed; pack K pairs -> b32 store ----
    {
      const float* r0 = Xb + (long)(kk + bKp) * xcs + (long)n0 * xps;
      const float* r1 = r0 + xcs;
      #pragma unroll
      for (int i = 0; i < 4; ++i) {
        int col = bC0 + i;
        unsigned int lo = f2bf(r0[(long)col * xps]);
        unsigned int hi = f2bf(r1[(long)col * xps]);
        *(unsigned int*)&Bs[col][bKp] = lo | (hi << 16);
      }
    }
    // prefetch next K panel (gfx1250 global_prefetch path)
    if (kk + 64 < Cin) {
      if (m0 + aRow < Cout)
        __builtin_prefetch(Wt + (long)(m0 + aRow) * Cin + kk + 64 + aK0, 0, 1);
      __builtin_prefetch(Xb + (long)(kk + 64 + bKp) * xcs + (long)(n0 + bC0) * xps, 0, 1);
    }
    __syncthreads();

    #pragma unroll
    for (int half = 0; half < 2; ++half) {
      const int kb = half * 32;
      // A fragment: lane<16 -> K {kb+klo..+7, kb+16+klo..+7}
      Bits256 aB;
      aB.lo = *(const uint4*)&As[rA][kb + kloA];
      aB.hi = *(const uint4*)&As[rA][kb + 16 + kloA];
      v16bf af = __builtin_bit_cast(v16bf, aB);
      // B fragment: lane owns col; K {kb+kb..+7, kb+16+kb..+7} (same as A)
      Bits256 bB;
      bB.lo = *(const uint4*)&Bs[colB][kb + kbB];
      bB.hi = *(const uint4*)&Bs[colB][kb + kbB + 16];
      v16bf bfv = __builtin_bit_cast(v16bf, bB);

      acc = __builtin_amdgcn_wmma_f32_16x16x32_bf16(
          /*neg_a=*/false, af, /*neg_b=*/false, bfv,
          /*c_mod=*/(short)0, acc, /*reuse_a=*/false, /*reuse_b=*/false);
    }
    __syncthreads();
  }

  // epilogue + store; C/D layout: VGPR j -> M = j + (lane>=16 ? 8 : 0)
  int colL  = wn * 16 + (lane & 15);
  int pix   = n0 + colL;
  int mBase = m0 + wm * 16 + ((lane >= 16) ? 8 : 0);
  #pragma unroll
  for (int j = 0; j < 8; ++j) {
    int m = mBase + j;
    if (m >= Cout) continue;
    float v = acc[j] + bias[m];
    if (epilogue == EP_GELU)      v = gelu_f(v);
    else if (epilogue == EP_GATE) v = v * (1.f / (1.f + expf(-gate[m])));
    long addr;
    if (vt_store) {
      addr = (long)batch * ((long)Cout * Npix)
           + (long)(m >> 5) * ((long)Npix * DD)
           + (long)pix * DD + (m & 31);
    } else {
      addr = (long)batch * ybs + (long)m * yos + (long)pix * yps;
    }
    Y[addr] = v;
  }
}

// ---------------------------------------------------------------------------
// Depthwise 3x3 (SAME, zero pad) + bias + GELU. grid = B*C blocks.
// ---------------------------------------------------------------------------
__global__ __launch_bounds__(256)
void dwconv3_gelu_kernel(const float* __restrict__ X, const float* __restrict__ Wd,
                         const float* __restrict__ bias, float* __restrict__ Y)
{
  int bc = blockIdx.x;
  int c  = bc & (CC - 1);
  const float* src = X + (long)bc * HWp;
  float*       dst = Y + (long)bc * HWp;
  float w[9];
  #pragma unroll
  for (int i = 0; i < 9; ++i) w[i] = Wd[c * 9 + i];
  float bv = bias[c];
  for (int p = threadIdx.x; p < HWp; p += 256) {
    int h = p >> 6, x = p & 63;
    float acc = bv;
    #pragma unroll
    for (int ky = 0; ky < 3; ++ky) {
      int yy = h + ky - 1;
      if (yy < 0 || yy > 63) continue;
      #pragma unroll
      for (int kx = 0; kx < 3; ++kx) {
        int xx = x + kx - 1;
        if (xx < 0 || xx > 63) continue;
        acc += w[ky * 3 + kx] * src[yy * WW + xx];
      }
    }
    dst[p] = gelu_f(acc);
  }
}

// ---------------------------------------------------------------------------
// Dense 3x3 conv (SAME, zero pad) + bias + GELU. One 16x16 pixel tile / block.
// grid = (16 tiles, Cout, B)
// ---------------------------------------------------------------------------
__global__ __launch_bounds__(256)
void conv3_gelu_kernel(const float* __restrict__ X, const float* __restrict__ Wc,
                       const float* __restrict__ bias, float* __restrict__ Y,
                       int Cin, int Cout)
{
  __shared__ float tile[18][18];
  int t  = blockIdx.x;
  int o  = blockIdx.y;
  int b  = blockIdx.z;
  int ty0 = (t >> 2) * 16, tx0 = (t & 3) * 16;
  int tx = threadIdx.x & 15, ty = threadIdx.x >> 4;
  float acc = bias[o];
  for (int ci = 0; ci < Cin; ++ci) {
    __syncthreads();
    const float* src = X + ((long)b * Cin + ci) * HWp;
    for (int e = threadIdx.x; e < 324; e += 256) {
      int r = e / 18, c = e % 18;
      int yy = ty0 - 1 + r, xx = tx0 - 1 + c;
      tile[r][c] = (yy >= 0 && yy < HH && xx >= 0 && xx < WW) ? src[yy * WW + xx] : 0.f;
    }
    __syncthreads();
    const float* wp = Wc + ((long)o * Cin + ci) * 9;
    #pragma unroll
    for (int ky = 0; ky < 3; ++ky)
      #pragma unroll
      for (int kx = 0; kx < 3; ++kx)
        acc += wp[ky * 3 + kx] * tile[ty + ky][tx + kx];
  }
  Y[((long)b * Cout + o) * HWp + (ty0 + ty) * WW + (tx0 + tx)] = gelu_f(acc);
}

// ---------------------------------------------------------------------------
// Time-encoding projections: off_scale (B,N,48), bias_pen (B,N,K), |dt| (B,N)
// ---------------------------------------------------------------------------
__global__ void scales_kernel(const float* __restrict__ te,
                              const float* __restrict__ tosw, const float* __restrict__ tosb,
                              const float* __restrict__ tabw, const float* __restrict__ tabb,
                              const float* __restrict__ rt,
                              float* __restrict__ off_scale, float* __restrict__ bias_pen,
                              float* __restrict__ dtv)
{
  int bn = blockIdx.x;             // 0..B*N-1
  int t  = threadIdx.x;            // 64
  const float* e = te + bn * TT;
  if (t < KK * PP) {
    float s = tosb[t];
    for (int i = 0; i < TT; ++i) s += e[i] * tosw[t * TT + i];
    float sp = (s > 20.f) ? s : log1pf(expf(s));      // softplus
    off_scale[bn * (KK * PP) + t] = 0.5f + sp;
  }
  if (t < KK) {
    float s = tabb[t];
    for (int i = 0; i < TT; ++i) s += e[i] * tabw[t * TT + i];
    bias_pen[bn * KK + t] = s - 2.f * fabsf(rt[bn]);
  }
  if (t == 0) dtv[bn] = fabsf(rt[bn]);
}

// ---------------------------------------------------------------------------
// Fused softmax + bilinear sampling + weighted sum.
// One wave32 per (b,k,pix); lane = d channel. Lanes 0..23 own (n,p) logits.
// ---------------------------------------------------------------------------
__global__ __launch_bounds__(256)
void deform_attn_kernel(const float* __restrict__ a3, const float* __restrict__ offm,
                        const float* __restrict__ off_scale, const float* __restrict__ bias_pen,
                        const float* __restrict__ dtv, const float* __restrict__ vt,
                        float* __restrict__ out_t, float* __restrict__ ent)
{
  const int lane = threadIdx.x & 31;
  const int wave = threadIdx.x >> 5;
  int gid = blockIdx.x * 8 + wave;          // < B*K*HW = 65536
  int b   = gid >> 15;
  int rem = gid & 32767;
  int k   = rem >> 12;
  int pix = rem & 4095;
  int h   = pix >> 6, wq = pix & 63;

  float logit = -3.0e38f;
  float wxv = 0.f, wyv = 0.f;
  int x0 = 0, x1 = 0, y0 = 0, y1 = 0;
  if (lane < NN * PP) {
    int n  = lane / PP, p = lane - n * PP;
    int kp = k * PP + p;
    logit = a3[((long)b * (KK * PP) + kp) * HWp + pix] + bias_pen[(b * NN + n) * KK + k];
    float ox = offm[((long)b * (KK * PP * 2) + kp * 2 + 0) * HWp + pix];
    float oy = offm[((long)b * (KK * PP * 2) + kp * 2 + 1) * HWp + pix];
    float f  = off_scale[(b * NN + n) * (KK * PP) + kp] * (1.f + dtv[b * NN + n]);
    float sx = fminf(fmaxf(ox * f, -1.f), 1.f);
    float sy = fminf(fmaxf(oy * f, -1.f), 1.f);
    float gx = fminf(fmaxf((-1.f + 2.f * wq / 63.f) + sx, -1.f), 1.f);
    float gy = fminf(fmaxf((-1.f + 2.f * h  / 63.f) + sy, -1.f), 1.f);
    float gxp = (gx + 1.f) * 0.5f * 63.f;
    float gyp = (gy + 1.f) * 0.5f * 63.f;
    float xf = floorf(gxp), yf = floorf(gyp);
    wxv = gxp - xf; wyv = gyp - yf;
    int xi = (int)xf, yi = (int)yf;
    x0 = min(max(xi, 0), 63);     x1 = min(max(xi + 1, 0), 63);
    y0 = min(max(yi, 0), 63);     y1 = min(max(yi + 1, 0), 63);
  }

  // wave-wide softmax over 24 valid lanes
  float m = logit;
  for (int s = 16; s; s >>= 1) m = fmaxf(m, __shfl_xor(m, s, 32));
  float e = (lane < NN * PP) ? expf(logit - m) : 0.f;
  float sum = e;
  for (int s = 16; s; s >>= 1) sum += __shfl_xor(sum, s, 32);
  float wgt = e / sum;
  float ev = (lane < NN * PP) ? -wgt * logf(wgt + 1e-8f) : 0.f;
  float esum = ev;
  for (int s = 16; s; s >>= 1) esum += __shfl_xor(esum, s, 32);
  if (lane == 0) atomicAdd(&ent[b * HWp + pix], esum * (1.f / KK));

  // weighted bilinear gather-accumulate; lane = d channel (coalesced)
  float acc = 0.f;
  #pragma unroll
  for (int j = 0; j < NN * PP; ++j) {
    float wj  = __shfl(wgt, j, 32);
    float wxj = __shfl(wxv, j, 32);
    float wyj = __shfl(wyv, j, 32);
    int x0j = __shfl(x0, j, 32);
    int x1j = __shfl(x1, j, 32);
    int y0j = __shfl(y0, j, 32);
    int y1j = __shfl(y1, j, 32);
    int nj = j / PP;
    const float* vb_ = vt + (((long)(b * NN + nj) * KK + k) * HWp) * DD;
    float v00 = vb_[(y0j * WW + x0j) * DD + lane];
    float v01 = vb_[(y0j * WW + x1j) * DD + lane];
    float v10 = vb_[(y1j * WW + x0j) * DD + lane];
    float v11 = vb_[(y1j * WW + x1j) * DD + lane];
    float bil = v00 * (1.f - wxj) * (1.f - wyj) + v01 * wxj * (1.f - wyj)
              + v10 * (1.f - wxj) * wyj         + v11 * wxj * wyj;
    acc += wj * bil;
  }
  // channel-last store (b, pix, c): out1 GEMM reads with strides (xcs=1, xps=C)
  out_t[((long)b * HWp + pix) * CC + k * DD + lane] = acc;
}

// ---------------------------------------------------------------------------
// confidence = 1 - clip(entropy/log(N*P), 0, 1); also emit entropy.
// ---------------------------------------------------------------------------
__global__ void conf_kernel(const float* __restrict__ ent,
                            float* __restrict__ outc, float* __restrict__ oute)
{
  int i = blockIdx.x * 256 + threadIdx.x;   // < B*HW
  float e = ent[i];
  outc[i] = 1.f - fminf(fmaxf(e / 3.17805383f, 0.f), 1.f);  // log(24)
  oute[i] = e;
}

// ---------------------------------------------------------------------------
// Host orchestration
// ---------------------------------------------------------------------------
static inline void launch_gemm(hipStream_t s,
                               const float* X, const float* Wt, const float* bias,
                               const float* gate, float* Y,
                               int Cout, int Cin, int Npix, int nbatch,
                               long xbs, long xcs, long xps,
                               long ybs, long yos, long yps,
                               int ep, int vt)
{
  dim3 grid((Cout + 63) / 64, Npix / 32, nbatch);
  gemm1x1_wmma<<<grid, 256, 0, s>>>(X, Wt, bias, gate, Y, Cout, Cin, Npix,
                                    xbs, xcs, xps, ybs, yos, yps, ep, vt);
}

extern "C" void kernel_launch(void* const* d_in, const int* in_sizes, int n_in,
                              void* d_out, int out_size, void* d_ws, size_t ws_size,
                              hipStream_t stream)
{
  (void)in_sizes; (void)n_in; (void)out_size; (void)ws_size;
  const float* query    = (const float*)d_in[0];
  const float* values   = (const float*)d_in[1];
  const float* rel_time = (const float*)d_in[2];
  const float* time_enc = (const float*)d_in[3];
  const float* qw  = (const float*)d_in[4];   const float* qb  = (const float*)d_in[5];
  const float* dww = (const float*)d_in[6];   const float* dwb = (const float*)d_in[7];
  const float* o1w = (const float*)d_in[8];   const float* o1b = (const float*)d_in[9];
  const float* o2w = (const float*)d_in[10];  const float* o2b = (const float*)d_in[11];
  const float* o3w = (const float*)d_in[12];  const float* o3b = (const float*)d_in[13];
  const float* tosw= (const float*)d_in[14];  const float* tosb= (const float*)d_in[15];
  const float* a1w = (const float*)d_in[16];  const float* a1b = (const float*)d_in[17];
  const float* a2w = (const float*)d_in[18];  const float* a2b = (const float*)d_in[19];
  const float* a3w = (const float*)d_in[20];  const float* a3b = (const float*)d_in[21];
  const float* tabw= (const float*)d_in[22];  const float* tabb= (const float*)d_in[23];
  const float* vw  = (const float*)d_in[24];  const float* vb  = (const float*)d_in[25];
  const float* u1w = (const float*)d_in[26];  const float* u1b = (const float*)d_in[27];
  const float* u2w = (const float*)d_in[28];  const float* u2b = (const float*)d_in[29];
  const float* gate= (const float*)d_in[30];

  float* ws = (float*)d_ws;
  // workspace layout (float offsets)
  float* Q    = ws + 0;            // B*C*HW  = 2097152
  float* bufA = ws + 2097152;      // 2097152 (dw out / attn1 / out1)
  float* bufB = ws + 4194304;      // 2097152 (off1 out / out_t)
  float* bufC = ws + 6291456;      // 1048576 (off2 out)
  float* OFF  = ws + 7340032;      //  786432 (off3 out)
  float* C2   = ws + 8126464;      //  524288 (attn2 out)
  float* ATT  = ws + 8650752;      //  393216 (attn3 out)
  float* VT   = ws + 9043968;      // 8388608 (channel-last v)
  float* ENT  = ws + 17432576;     //    8192
  float* SCL  = ws + 17440768;     //     384
  float* BP   = ws + 17441152;     //      64
  float* DT   = ws + 17441216;     //       8

  float* out_main = (float*)d_out;                       // (B,C,H,W)
  float* out_conf = out_main + (long)BB * CC * HWp;      // (B,1,H,W)
  float* out_entr = out_conf + (long)BB * HWp;           // (B,1,H,W)

  const long CS = (long)CC * HWp;   // NCHW per-batch stride for C=256

  // 1) q = conv1x1(query, qw) + qb
  launch_gemm(stream, query, qw, qb, nullptr, Q, CC, CC, HWp, BB,
              CS, HWp, 1, CS, HWp, 1, EP_NONE, 0);
  // 2) x = gelu(dwconv3(q))
  dwconv3_gelu_kernel<<<BB * CC, 256, 0, stream>>>(Q, dww, dwb, bufA);
  // 3) x = gelu(conv1x1(x, off1))
  launch_gemm(stream, bufA, o1w, o1b, nullptr, bufB, CC, CC, HWp, BB,
              CS, HWp, 1, CS, HWp, 1, EP_GELU, 0);
  // 4) x = gelu(conv1x1(x, off2))  C -> C/2
  launch_gemm(stream, bufB, o2w, o2b, nullptr, bufC, CC / 2, CC, HWp, BB,
              CS, HWp, 1, (long)(CC / 2) * HWp, HWp, 1, EP_GELU, 0);
  // 5) off = conv1x1(x, off3)  C/2 -> 96
  launch_gemm(stream, bufC, o3w, o3b, nullptr, OFF, KK * PP * 2, CC / 2, HWp, BB,
              (long)(CC / 2) * HWp, HWp, 1, (long)(KK * PP * 2) * HWp, HWp, 1, EP_NONE, 0);
  // 6) a = gelu(conv1x1(q, attn1))  C -> C/2
  launch_gemm(stream, Q, a1w, a1b, nullptr, bufA, CC / 2, CC, HWp, BB,
              CS, HWp, 1, (long)(CC / 2) * HWp, HWp, 1, EP_GELU, 0);
  // 7) a = gelu(conv3(a, attn2))  C/2 -> C/4
  conv3_gelu_kernel<<<dim3(16, CC / 4, BB), 256, 0, stream>>>(bufA, a2w, a2b, C2,
                                                              CC / 2, CC / 4);
  // 8) a3 = conv1x1(a, attn3)  C/4 -> 48
  launch_gemm(stream, C2, a3w, a3b, nullptr, ATT, KK * PP, CC / 4, HWp, BB,
              (long)(CC / 4) * HWp, HWp, 1, (long)(KK * PP) * HWp, HWp, 1, EP_NONE, 0);
  // 9) v = conv1x1(values, vw), stored channel-last (bn,k,pix,d)
  launch_gemm(stream, values, vw, vb, nullptr, VT, CC, CC, HWp, BB * NN,
              CS, HWp, 1, 0, 0, 0, EP_NONE, 1);
  // 10) time projections
  scales_kernel<<<BB * NN, 64, 0, stream>>>(time_enc, tosw, tosb, tabw, tabb, rel_time,
                                            SCL, BP, DT);
  // 11) fused softmax + deformable sampling + weighted sum
  hipMemsetAsync(ENT, 0, (size_t)BB * HWp * sizeof(float), stream);
  deform_attn_kernel<<<(BB * KK * HWp) / 8, 256, 0, stream>>>(ATT, OFF, SCL, BP, DT, VT,
                                                              bufB, ENT);
  // 12) out = gelu(conv1x1(out_t, out1))  — input is channel-last (xcs=1, xps=C)
  launch_gemm(stream, bufB, u1w, u1b, nullptr, bufA, CC, CC, HWp, BB,
              (long)HWp * CC, 1, CC, CS, HWp, 1, EP_GELU, 0);
  // 13) out = conv1x1(out, out2) * sigmoid(gate)
  launch_gemm(stream, bufA, u2w, u2b, gate, out_main, CC, CC, HWp, BB,
              CS, HWp, 1, CS, HWp, 1, EP_GATE, 0);
  // 14) confidence / entropy outputs
  conf_kernel<<<(BB * HWp) / 256, 256, 0, stream>>>(ENT, out_conf, out_entr);
}
